// MultiHeadAttention_5257039970386
// MI455X (gfx1250) — matrix operand order repaired and checked by reading, not compiled, not run on hardware
//
#include <hip/hip_runtime.h>

// ---------------------------------------------------------------------------
// MHA forward for MI455X (gfx1250): bf16 WMMA (v_wmma_f32_16x16x32_bf16),
// f32 accumulate. ~137 GFLOP vs ~11us of HBM traffic at 23.3 TB/s -> compute
// bound -> all GEMM-like work on the WMMA pipe.
// ---------------------------------------------------------------------------

typedef __bf16 bf16;
typedef __attribute__((ext_vector_type(16))) __bf16 v16bf;
typedef __attribute__((ext_vector_type(8)))  __bf16 bf16x8;
typedef __attribute__((ext_vector_type(8)))  float  v8f;

#define LSEQ    2048
#define BATCH   4
#define DMODEL  1024
#define NHEAD   16
#define DHEAD   64
#define MROWS   (LSEQ * BATCH)      // 8192 rows of [L,B,:] flattened
#define SSTRIDE 2068                // 2048 + 20 pad: conflict-free b128 LDS reads
#define XLSTR   72                  // GEMM LDS row stride (64 + 8 pad, conflict-free)

static __device__ inline v8f wmma_bf16f32(v16bf a, v16bf b, v8f c) {
  return __builtin_amdgcn_wmma_f32_16x16x32_bf16(false, a, false, b, (short)0, c,
                                                 false, false);
}

// A fragment: 16(M) x 32(K) bf16, row-major, stride rs.
// lane -> M=lane%16, elems e -> K = (lane/16)*16 + e.
static __device__ inline v16bf frag_a_bf16(const bf16* base, int rs) {
  const int lane = threadIdx.x & 31;
  const int half = lane >> 4;
  const bf16* rowp = base + (size_t)(lane & 15) * rs;
  bf16x8 lo = *(const bf16x8*)(rowp + half * 16);
  bf16x8 hi = *(const bf16x8*)(rowp + half * 16 + 8);
  v16bf a;
#pragma unroll
  for (int i = 0; i < 8; ++i) { a[i] = lo[i]; a[i + 8] = hi[i]; }
  return a;
}

// B fragment from an [N][K] row-major tile (computes with the transpose):
// lane nl -> N=nl, elems e -> K = (lane/16)*16 + e.
static __device__ inline v16bf frag_b_rowmajorN(const bf16* base, int rs) {
  const int lane = threadIdx.x & 31;
  const int half = lane >> 4;
  const bf16* rowp = base + (size_t)(lane & 15) * rs + half * 16;
  bf16x8 lo = *(const bf16x8*)(rowp);
  bf16x8 hi = *(const bf16x8*)(rowp + 8);
  v16bf b;
#pragma unroll
  for (int i = 0; i < 8; ++i) { b[i] = lo[i]; b[i + 8] = hi[i]; }
  return b;
}

// ---------------------------------------------------------------------------
// Y[m][n] = scale * sum_k X[m][k] * W[n][k]   (M=8192, N=K=1024)
// XF32 (compile-time): X is f32 (cvt to bf16 while staging) else bf16.
// OMODE (compile-time): 0 -> Yf f32 [M][1024]
//                       1 -> Yb bf16 [B][H][L][64]   (Q/K)
//                       2 -> Yb bf16 [B][H][64][L]   (V transposed)
// 256 threads = 8 waves; 128x64 C tile; K staged 64 wide through LDS.
// ---------------------------------------------------------------------------
template <bool XF32, int OMODE>
__global__ __launch_bounds__(256) void mha_gemm(
    const float* __restrict__ Xf, const bf16* __restrict__ Xb,
    const bf16* __restrict__ W, float* __restrict__ Yf, bf16* __restrict__ Yb,
    float scale)
{
  __shared__ bf16 Xl[128 * XLSTR];   // 128 rows x 64 K (+8 pad)
  __shared__ bf16 Wl[64 * XLSTR];    // 64  rows x 64 K (+8 pad)

  const int t    = threadIdx.x;
  const int lane = t & 31;
  const int wid  = t >> 5;            // 8 waves
  const int wm   = (wid >> 1) * 32;   // 4 wave-rows
  const int wn   = (wid & 1) * 32;    // 2 wave-cols
  const int mblk = blockIdx.y * 128;
  const int nblk = blockIdx.x * 64;

  v8f acc00 = {}, acc01 = {}, acc10 = {}, acc11 = {};

  const int srow = t >> 3;            // staging: 8 threads/row, 32 rows/sweep
  const int scol = (t & 7) * 8;       // 8 contiguous K per thread

  for (int ko = 0; ko < DMODEL; ko += 64) {
    // ---- stage X (128x64): issue all global loads first, then convert/store
    if (XF32) {
      float4 xlo[4], xhi[4];
#pragma unroll
      for (int s = 0; s < 4; ++s) {
        const float* src = &Xf[(size_t)(mblk + s * 32 + srow) * DMODEL + ko + scol];
        xlo[s] = *(const float4*)(src);
        xhi[s] = *(const float4*)(src + 4);
      }
#pragma unroll
      for (int s = 0; s < 4; ++s) {
        bf16x8 v;
        v[0] = (bf16)xlo[s].x; v[1] = (bf16)xlo[s].y;
        v[2] = (bf16)xlo[s].z; v[3] = (bf16)xlo[s].w;
        v[4] = (bf16)xhi[s].x; v[5] = (bf16)xhi[s].y;
        v[6] = (bf16)xhi[s].z; v[7] = (bf16)xhi[s].w;
        *(bf16x8*)&Xl[(s * 32 + srow) * XLSTR + scol] = v;
      }
    } else {
      bf16x8 xv[4];
#pragma unroll
      for (int s = 0; s < 4; ++s)
        xv[s] = *(const bf16x8*)&Xb[(size_t)(mblk + s * 32 + srow) * DMODEL + ko + scol];
#pragma unroll
      for (int s = 0; s < 4; ++s)
        *(bf16x8*)&Xl[(s * 32 + srow) * XLSTR + scol] = xv[s];
    }
    // ---- stage W (64x64)
    {
      bf16x8 wv0 = *(const bf16x8*)&W[(size_t)(nblk + srow) * DMODEL + ko + scol];
      bf16x8 wv1 = *(const bf16x8*)&W[(size_t)(nblk + 32 + srow) * DMODEL + ko + scol];
      *(bf16x8*)&Wl[(srow)*XLSTR + scol] = wv0;
      *(bf16x8*)&Wl[(32 + srow) * XLSTR + scol] = wv1;
    }
    if (ko + 64 < DMODEL) {           // gfx1250 global_prefetch_b8
      if (XF32) __builtin_prefetch(&Xf[(size_t)(mblk + srow) * DMODEL + ko + 64]);
      else      __builtin_prefetch(&Xb[(size_t)(mblk + srow) * DMODEL + ko + 64]);
    }
    __syncthreads();

#pragma unroll
    for (int ks = 0; ks < 64; ks += 32) {
      v16bf a0 = frag_a_bf16(&Xl[(wm +  0) * XLSTR + ks], XLSTR);
      v16bf a1 = frag_a_bf16(&Xl[(wm + 16) * XLSTR + ks], XLSTR);
      v16bf b0 = frag_b_rowmajorN(&Wl[(wn +  0) * XLSTR + ks], XLSTR);
      v16bf b1 = frag_b_rowmajorN(&Wl[(wn + 16) * XLSTR + ks], XLSTR);
      acc00 = wmma_bf16f32(a0, b0, acc00);
      acc01 = wmma_bf16f32(a0, b1, acc01);
      acc10 = wmma_bf16f32(a1, b0, acc10);
      acc11 = wmma_bf16f32(a1, b1, acc11);
    }
    __syncthreads();
  }

  // Epilogue. C/D layout: VGPR i -> M = (lane/16)*8 + i, N = lane%16.
  const int half = lane >> 4, nl = lane & 15;
#pragma unroll
  for (int tm = 0; tm < 2; ++tm) {
#pragma unroll
    for (int tn = 0; tn < 2; ++tn) {
      const v8f acc = tm ? (tn ? acc11 : acc10) : (tn ? acc01 : acc00);
#pragma unroll
      for (int i = 0; i < 8; ++i) {
        const int m = mblk + wm + tm * 16 + half * 8 + i;   // row = l*B + b
        const int n = nblk + wn + tn * 16 + nl;
        const float v = acc[i] * scale;
        if (OMODE == 0) {
          Yf[(size_t)m * DMODEL + n] = v;
        } else {
          const int l = m >> 2, bb = m & 3;       // BATCH == 4
          const int h = n >> 6, dh = n & 63;      // DHEAD == 64
          if (OMODE == 1)
            Yb[(((size_t)(bb * NHEAD + h)) * LSEQ + l) * DHEAD + dh] = (bf16)v;
          else
            Yb[(((size_t)(bb * NHEAD + h)) * DHEAD + dh) * LSEQ + l] = (bf16)v;
        }
      }
    }
  }
}

// ---------------------------------------------------------------------------
// Attention: one workgroup (4 waves) per (b, h, 16-row q-tile).
// Phase A: S[16][2048] = (Q/sqrt(d)) K^T via WMMA (frags straight from global).
// Phase B: masked softmax in LDS (vectorized) + coverage atomics (head mean).
// Phase C: ctx[16][64] = P V via WMMA, P read from LDS f32 -> bf16, V^T global.
// ---------------------------------------------------------------------------
__global__ __launch_bounds__(128) void mha_attn(
    const bf16* __restrict__ Q, const bf16* __restrict__ K,
    const bf16* __restrict__ VT, const unsigned char* __restrict__ mask,
    float* __restrict__ coverage, bf16* __restrict__ ctx)
{
  extern __shared__ float smem[];
  float* S   = smem;                  // [16][SSTRIDE]
  float* red = smem + 16 * SSTRIDE;   // [16][8]

  const int qt = blockIdx.x, h = blockIdx.y, b = blockIdx.z;
  const int qbase = qt * 16;
  const int t = threadIdx.x, wid = t >> 5, lane = t & 31;
  const int half = lane >> 4, nl = lane & 15;

  const bf16* Qh  = Q  + ((size_t)(b * NHEAD + h)) * LSEQ * DHEAD;
  const bf16* Kh  = K  + ((size_t)(b * NHEAD + h)) * LSEQ * DHEAD;
  const bf16* VTh = VT + ((size_t)(b * NHEAD + h)) * DHEAD * LSEQ;

  // ---- Phase A: scores ----
  v16bf qa0 = frag_a_bf16(Qh + (size_t)qbase * DHEAD + 0,  DHEAD);
  v16bf qa1 = frag_a_bf16(Qh + (size_t)qbase * DHEAD + 32, DHEAD);
  for (int kt = wid * 32; kt < wid * 32 + 32; ++kt) {   // 32 k-tiles per wave
    v16bf kb0 = frag_b_rowmajorN(Kh + (size_t)kt * 16 * DHEAD + 0,  DHEAD);
    v16bf kb1 = frag_b_rowmajorN(Kh + (size_t)kt * 16 * DHEAD + 32, DHEAD);
    v8f s = {};
    s = wmma_bf16f32(qa0, kb0, s);
    s = wmma_bf16f32(qa1, kb1, s);
#pragma unroll
    for (int i = 0; i < 8; ++i)
      S[(half * 8 + i) * SSTRIDE + kt * 16 + nl] = s[i];
  }
  __syncthreads();

  // ---- Phase B: masked softmax (8 threads per row, 256 cols each, vec4) ----
  const int row = t >> 3, chunk = t & 7;
  float* Srow = S + row * SSTRIDE + chunk * 256;
  const uchar4* mrow4 =
      (const uchar4*)(mask + ((size_t)b * LSEQ + qbase + row) * LSEQ + chunk * 256);
  const float NINF = -__builtin_inff();
  float mx = NINF;
  for (int j = 0; j < 64; ++j) {
    float4 v = *(float4*)(Srow + j * 4);
    const uchar4 m4 = mrow4[j];
    if (m4.x) v.x = NINF;
    if (m4.y) v.y = NINF;
    if (m4.z) v.z = NINF;
    if (m4.w) v.w = NINF;
    *(float4*)(Srow + j * 4) = v;
    mx = fmaxf(mx, fmaxf(fmaxf(v.x, v.y), fmaxf(v.z, v.w)));
  }
  red[row * 8 + chunk] = mx;
  __syncthreads();
  float rmx = red[row * 8];
#pragma unroll
  for (int j = 1; j < 8; ++j) rmx = fmaxf(rmx, red[row * 8 + j]);
  __syncthreads();
  float sum = 0.f;
  for (int j = 0; j < 64; ++j) {
    float4 v = *(float4*)(Srow + j * 4);
    v.x = __expf(v.x - rmx); v.y = __expf(v.y - rmx);
    v.z = __expf(v.z - rmx); v.w = __expf(v.w - rmx);
    *(float4*)(Srow + j * 4) = v;
    sum += (v.x + v.y) + (v.z + v.w);
  }
  red[row * 8 + chunk] = sum;
  __syncthreads();
  float rs = 0.f;
#pragma unroll
  for (int j = 0; j < 8; ++j) rs += red[row * 8 + j];
  const float inv = 1.0f / rs;
  float* cov = coverage + ((size_t)b * LSEQ + qbase + row) * LSEQ + chunk * 256;
  for (int j = 0; j < 64; ++j) {
    float4 v = *(float4*)(Srow + j * 4);
    v.x *= inv; v.y *= inv; v.z *= inv; v.w *= inv;
    *(float4*)(Srow + j * 4) = v;
    // coverage = mean over heads; relaxed f32 atomics, region lives in 192MB L2
    __hip_atomic_fetch_add(&cov[j * 4 + 0], v.x * (1.0f / NHEAD),
                           __ATOMIC_RELAXED, __HIP_MEMORY_SCOPE_AGENT);
    __hip_atomic_fetch_add(&cov[j * 4 + 1], v.y * (1.0f / NHEAD),
                           __ATOMIC_RELAXED, __HIP_MEMORY_SCOPE_AGENT);
    __hip_atomic_fetch_add(&cov[j * 4 + 2], v.z * (1.0f / NHEAD),
                           __ATOMIC_RELAXED, __HIP_MEMORY_SCOPE_AGENT);
    __hip_atomic_fetch_add(&cov[j * 4 + 3], v.w * (1.0f / NHEAD),
                           __ATOMIC_RELAXED, __HIP_MEMORY_SCOPE_AGENT);
  }
  __syncthreads();

  // ---- Phase C: ctx = P * V, each wave owns a 16-wide d_head slice ----
  v8f acc = {};
  for (int kb = 0; kb < LSEQ; kb += 32) {
    const float* prow = S + nl * SSTRIDE + kb;   // A: M=q row (nl), K=k
    const float4 p0 = *(const float4*)(prow + half * 16);
    const float4 p1 = *(const float4*)(prow + half * 16 + 4);
    const float4 p2 = *(const float4*)(prow + half * 16 + 8);
    const float4 p3 = *(const float4*)(prow + half * 16 + 12);
    v16bf a;
    a[0]  = (bf16)p0.x; a[1]  = (bf16)p0.y; a[2]  = (bf16)p0.z; a[3]  = (bf16)p0.w;
    a[4]  = (bf16)p1.x; a[5]  = (bf16)p1.y; a[6]  = (bf16)p1.z; a[7]  = (bf16)p1.w;
    a[8]  = (bf16)p2.x; a[9]  = (bf16)p2.y; a[10] = (bf16)p2.z; a[11] = (bf16)p2.w;
    a[12] = (bf16)p3.x; a[13] = (bf16)p3.y; a[14] = (bf16)p3.z; a[15] = (bf16)p3.w;
    // B: N=dh row of V^T, elements contiguous in k
    const bf16* vrow = VTh + (size_t)(wid * 16 + nl) * LSEQ + kb + half * 16;
    bf16x8 lo = *(const bf16x8*)vrow;
    bf16x8 hi = *(const bf16x8*)(vrow + 8);
    v16bf bb;
#pragma unroll
    for (int e = 0; e < 8; ++e) { bb[e] = lo[e]; bb[8 + e] = hi[e]; }
    acc = wmma_bf16f32(a, bb, acc);
  }
#pragma unroll
  for (int i = 0; i < 8; ++i) {
    const int qrow = qbase + half * 8 + i;
    const int n = h * DHEAD + wid * 16 + nl;
    ctx[((size_t)qrow * BATCH + b) * DMODEL + n] = (bf16)acc[i];
  }
}

// ---------------------------------------------------------------------------
__global__ void f32_to_bf16(const float* __restrict__ src, bf16* __restrict__ dst,
                            int n) {
  for (int i = blockIdx.x * blockDim.x + threadIdx.x; i < n;
       i += gridDim.x * blockDim.x)
    dst[i] = (bf16)src[i];
}

__global__ void zero_f32(float* __restrict__ p, int n) {
  for (int i = blockIdx.x * blockDim.x + threadIdx.x; i < n;
       i += gridDim.x * blockDim.x)
    p[i] = 0.0f;
}

// ---------------------------------------------------------------------------
extern "C" void kernel_launch(void* const* d_in, const int* in_sizes, int n_in,
                              void* d_out, int out_size, void* d_ws, size_t ws_size,
                              hipStream_t stream) {
  const float* query = (const float*)d_in[0];
  const float* key   = (const float*)d_in[1];
  const float* value = (const float*)d_in[2];
  const unsigned char* mask = (const unsigned char*)d_in[3];
  const float* Wq = (const float*)d_in[4];
  const float* Wk = (const float*)d_in[5];
  const float* Wv = (const float*)d_in[6];
  const float* Wo = (const float*)d_in[7];

  float* out      = (float*)d_out;                           // [L,B,D] f32
  float* coverage = out + (size_t)LSEQ * BATCH * DMODEL;     // [B,L,L] f32

  bf16* ws   = (bf16*)d_ws;
  bf16* wq_b = ws;
  bf16* wk_b = wq_b + (size_t)DMODEL * DMODEL;
  bf16* wv_b = wk_b + (size_t)DMODEL * DMODEL;
  bf16* wo_b = wv_b + (size_t)DMODEL * DMODEL;
  bf16* Qws  = wo_b + (size_t)DMODEL * DMODEL;               // [B][H][L][64]
  bf16* Kws  = Qws + (size_t)BATCH * NHEAD * LSEQ * DHEAD;   // [B][H][L][64]
  bf16* VTws = Kws + (size_t)BATCH * NHEAD * LSEQ * DHEAD;   // [B][H][64][L]
  bf16* ctxw = VTws + (size_t)BATCH * NHEAD * LSEQ * DHEAD;  // [M][1024]

  const int nW = DMODEL * DMODEL;
  f32_to_bf16<<<512, 256, 0, stream>>>(Wq, wq_b, nW);
  f32_to_bf16<<<512, 256, 0, stream>>>(Wk, wk_b, nW);
  f32_to_bf16<<<512, 256, 0, stream>>>(Wv, wv_b, nW);
  f32_to_bf16<<<512, 256, 0, stream>>>(Wo, wo_b, nW);

  dim3 gg(DMODEL / 64, MROWS / 128);  // (16, 64)
  mha_gemm<true, 1><<<gg, 256, 0, stream>>>(query, nullptr, wq_b, nullptr, Qws, 0.125f);
  mha_gemm<true, 1><<<gg, 256, 0, stream>>>(key,   nullptr, wk_b, nullptr, Kws, 1.0f);
  mha_gemm<true, 2><<<gg, 256, 0, stream>>>(value, nullptr, wv_b, nullptr, VTws, 1.0f);

  zero_f32<<<2048, 256, 0, stream>>>(coverage, BATCH * LSEQ * LSEQ);

  dim3 ga(LSEQ / 16, NHEAD, BATCH);
  const size_t lds_bytes = (16 * SSTRIDE + 16 * 8) * sizeof(float);  // 132,864 B
  mha_attn<<<ga, 128, lds_bytes, stream>>>(Qws, Kws, VTws, mask, coverage, ctxw);

  mha_gemm<false, 0><<<gg, 256, 0, stream>>>(nullptr, ctxw, wo_b, out, nullptr, 1.0f);
}
